// TransLayer_45217415692537
// MI455X (gfx1250) — matrix operand order repaired and checked by reading, not compiled, not run on hardware
//
#include <hip/hip_runtime.h>

typedef __bf16 bhalf;
typedef __attribute__((ext_vector_type(16))) __bf16 v16bf;
typedef __attribute__((ext_vector_type(8)))  __bf16 v8bf;
typedef __attribute__((ext_vector_type(8)))  float  v8f;
typedef __attribute__((ext_vector_type(2)))  float  v2f;
typedef __attribute__((ext_vector_type(4))) unsigned int u32x4;
typedef __attribute__((ext_vector_type(8))) int i32x8;
typedef __attribute__((ext_vector_type(4))) int i32x4;

#if __has_builtin(__builtin_amdgcn_wmma_f32_16x16x4_f32)
#define HAVE_WMMA_F32 1
#else
#define HAVE_WMMA_F32 0
#endif

#if __has_builtin(__builtin_amdgcn_tensor_load_to_lds)
#define HAVE_TDM 1
#else
#define HAVE_TDM 0
#endif

#if __has_include(<hip/amd_detail/amd_gfx1250_TDM.h>)
#define TDM_SIX_ARGS 1
#else
#define TDM_SIX_ARGS 0
#endif

#define NB 8
#define SEQ 4096
#define DMODEL 512
#define NH 8
#define DH 64
#define NL 256
#define BH (NB*NH)    /* 64  */
#define NT (NB*SEQ)   /* 32768 */
#define KERN 33

// ---------------------------------------------------------------------------
// WMMA fragment helpers (bf16, 16x16x32)
// A 16x32: lane<16 -> row M=lane, K {0..15};
//          lane>=16 -> K {16..31}.  Two contiguous 16B chunks / lane.
// B stored as Bt[N][K] (N-major): lane = N, 16 contiguous K elements / lane,
//          lanes 16..31 take K+16.
// C/D 16x16 f32: lane = N (mod 16), vgpr j -> M = j + 8*(lane>>4).
// ---------------------------------------------------------------------------
__device__ __forceinline__ v16bf load_a_frag(const bhalf* A, int lda, int row, int k0) {
  int lane = threadIdx.x & 31;
  const bhalf* p = A + (size_t)(row + (lane & 15)) * lda + k0 + ((lane >> 4) << 4);
  v8bf lo = *(const v8bf*)(p);
  v8bf hi = *(const v8bf*)(p + 8);
  v16bf r;
#pragma unroll
  for (int i = 0; i < 8; ++i) { r[i] = lo[i]; r[i + 8] = hi[i]; }
  return r;
}

__device__ __forceinline__ void store_tile(bhalf* Cb, float* Cf, int ldc, int row,
                                           int col, v8f c, float scale, int transC) {
  int lane = threadIdx.x & 31;
  int cc = col + (lane & 15);
  int r0 = row + ((lane >> 4) << 3);
#pragma unroll
  for (int j = 0; j < 8; ++j) {
    float v = c[j] * scale;
    size_t off = transC ? ((size_t)cc * ldc + (r0 + j))
                        : ((size_t)(r0 + j) * ldc + cc);
    if (Cb) Cb[off] = (bhalf)v; else Cf[off] = v;
  }
}

// ---------------------------------------------------------------------------
// Tensor Data Mover: async DMA of a 64-row x 64-K bf16 tile (8 KB) from a
// row-major [rows, K] bf16 tensor into LDS (rows packed at 128 B stride).
// D# built per CDNA5 ISA ch.8 (2D, data_size=2B, type=2).
// ---------------------------------------------------------------------------
#if HAVE_TDM
__device__ __forceinline__ void tdm_load_tile_64x64(const bhalf* gsrc,
                                                    unsigned lds_byte_off, int K) {
  unsigned long long ga = (unsigned long long)(const void*)gsrc;
  unsigned dim0 = (unsigned)K;   // tensor_dim0 (elements, K-contiguous)
  unsigned dim1 = 64u;           // rows available from tile origin
  u32x4 g0;
  g0[0] = 1u;                                                   // count=1 (valid)
  g0[1] = lds_byte_off;                                         // lds_addr
  g0[2] = (unsigned)ga;                                         // global_addr lo
  g0[3] = (unsigned)((ga >> 32) & 0x01ffffffull) | 0x80000000u; // addr hi | type=2
  i32x8 g1;
  g1[0] = 0x00010000;                                    // data_size=1 -> 2 bytes
  g1[1] = (int)((dim0 & 0xffffu) << 16);                 // tensor_dim0 lo16
  g1[2] = (int)(((dim0 >> 16) & 0xffffu) | ((dim1 & 0xffffu) << 16));
  g1[3] = (int)(((dim1 >> 16) & 0xffffu) | (64u << 16)); // tile_dim0 = 64
  g1[4] = 64;                                            // tile_dim1=64, tile_dim2=0
  g1[5] = (int)(unsigned)K;                              // tensor_dim0_stride lo32
  g1[6] = 0;                                             // stride hi16 | dim1_stride lo
  g1[7] = 0;
  i32x4 g2; g2[0] = 1; g2[1] = 1; g2[2] = 0; g2[3] = 0;  // tensor_dim2=1, dim3=1
  i32x4 g3; g3[0] = 0; g3[1] = (int)(1u << 16); g3[2] = 0; g3[3] = 0; // dim4=1
#if TDM_SIX_ARGS
  i32x8 gx = {};
  __builtin_amdgcn_tensor_load_to_lds(g0, g1, g2, g3, gx, 0);
#else
  __builtin_amdgcn_tensor_load_to_lds(g0, g1, g2, g3, 0);
#endif
}
#endif

// ---------------------------------------------------------------------------
// Generic bf16 GEMM:  C[M,N] = scale * A[M,K] x Bt[N,K]^T   (row-major A, Bt)
// wave tile 64x64, block = 4 waves (256 rows); B tile (shared by all 4 waves)
// staged in LDS by the TDM, double-buffered; batched over blockIdx.z.
// Requires M % 256 == 0, N % 64 == 0, K % 64 == 0.
// ---------------------------------------------------------------------------
__global__ void __launch_bounds__(128)
gemm_bf16_abT(const bhalf* A, long long sA, const bhalf* Bt, long long sB,
              bhalf* Cb, float* Cf, long long sC,
              int M, int N, int K, int transC, float scale) {
  __shared__ alignas(128) bhalf btile[2][64 * 64];
  int bz = blockIdx.z;
  A  += (size_t)bz * sA;
  Bt += (size_t)bz * sB;
  int wave = threadIdx.x >> 5;
  int lane = threadIdx.x & 31;
  int row0 = blockIdx.x * 256 + wave * 64;
  int col0 = blockIdx.y * 64;
  const bhalf* bsrc = Bt + (size_t)col0 * K;   // origin of this block's B panel
  int nk = K >> 6;

  v8f acc[4][4] = {};

#if HAVE_TDM
  unsigned ldsaddr0 = (unsigned)(size_t)(void*)&btile[0][0];
  unsigned ldsaddr1 = (unsigned)(size_t)(void*)&btile[1][0];
  if (wave == 0) {
    tdm_load_tile_64x64(bsrc, ldsaddr0, K);
    __builtin_amdgcn_s_wait_tensorcnt(0);
  }
  __syncthreads();
  for (int t = 0; t < nk; ++t) {
    if (wave == 0 && (t + 1) < nk)
      tdm_load_tile_64x64(bsrc + ((size_t)(t + 1) << 6),
                          ((t + 1) & 1) ? ldsaddr1 : ldsaddr0, K);
    int buf = t & 1;
    int k0 = t << 6;
#pragma unroll
    for (int ks = 0; ks < 64; ks += 32) {
      v16bf a[4], bfr[4];
#pragma unroll
      for (int i = 0; i < 4; ++i) a[i] = load_a_frag(A, K, row0 + 16 * i, k0 + ks);
#pragma unroll
      for (int j = 0; j < 4; ++j)
        bfr[j] = *(const v16bf*)&btile[buf][(size_t)(16 * j + (lane & 15)) * 64 +
                                            ks + ((lane >> 4) << 4)];
#pragma unroll
      for (int j = 0; j < 4; ++j)
#pragma unroll
        for (int i = 0; i < 4; ++i)
          acc[i][j] = __builtin_amdgcn_wmma_f32_16x16x32_bf16(false, a[i], false, bfr[j],
                                                              (short)0, acc[i][j], false, false);
    }
    if (wave == 0) __builtin_amdgcn_s_wait_tensorcnt(0);
    __syncthreads();
  }
#else
  for (int t = 0; t < nk; ++t) {
    int k0 = t << 6;
    __syncthreads();
    for (int c = threadIdx.x; c < 512; c += 128) {  // 64 rows x 64 elems, 8-elem chunks
      int r = c >> 3, cc = (c & 7) << 3;
      *(v8bf*)&btile[0][(size_t)r * 64 + cc] = *(const v8bf*)(bsrc + (size_t)r * K + k0 + cc);
    }
    __syncthreads();
#pragma unroll
    for (int ks = 0; ks < 64; ks += 32) {
      v16bf a[4], bfr[4];
#pragma unroll
      for (int i = 0; i < 4; ++i) a[i] = load_a_frag(A, K, row0 + 16 * i, k0 + ks);
#pragma unroll
      for (int j = 0; j < 4; ++j)
        bfr[j] = *(const v16bf*)&btile[0][(size_t)(16 * j + (lane & 15)) * 64 +
                                          ks + ((lane >> 4) << 4)];
#pragma unroll
      for (int j = 0; j < 4; ++j)
#pragma unroll
        for (int i = 0; i < 4; ++i)
          acc[i][j] = __builtin_amdgcn_wmma_f32_16x16x32_bf16(false, a[i], false, bfr[j],
                                                              (short)0, acc[i][j], false, false);
    }
  }
#endif

  if (Cb) Cb += (size_t)bz * sC;
  if (Cf) Cf += (size_t)bz * sC;
  int ldc = transC ? M : N;
#pragma unroll
  for (int i = 0; i < 4; ++i)
#pragma unroll
    for (int j = 0; j < 4; ++j)
      store_tile(Cb, Cf, ldc, row0 + 16 * i, col0 + 16 * j, acc[i][j], scale, transC);
}

// ---------------------------------------------------------------------------
// fp32 GEMM for the Newton-Schulz pinv: C = scale * A[256,256] x B[256,256]
// (both row-major, batch 64 at stride 65536), V_WMMA_F32_16X16X4_F32
// ---------------------------------------------------------------------------
__global__ void __launch_bounds__(128)
gemm_f32_ab(const float* A, const float* B, float* C, int M, int N, int K, float scale) {
  size_t bz = blockIdx.z;
  A += bz * 65536; B += bz * 65536; C += bz * 65536;
  int wave = threadIdx.x >> 5;
  int row0 = blockIdx.x * 128 + wave * 32;
  int col0 = blockIdx.y * 64;
  if (row0 >= M) return;
#if HAVE_WMMA_F32
  int lane = threadIdx.x & 31;
  int kh = (lane >> 4) << 1;   // K sub-offset 0 or 2
  v8f acc[2][4] = {};
  for (int k0 = 0; k0 < K; k0 += 4) {
    v2f a0, a1, bb[4];
    {
      const float* p0 = A + (size_t)(row0 + (lane & 15)) * K + k0 + kh;
      const float* p1 = p0 + (size_t)16 * K;
      a0[0] = p0[0]; a0[1] = p0[1];
      a1[0] = p1[0]; a1[1] = p1[1];
    }
#pragma unroll
    for (int j = 0; j < 4; ++j) {
      const float* p = B + (size_t)(k0 + kh) * N + col0 + 16 * j + (lane & 15);
      bb[j][0] = p[0]; bb[j][1] = p[N];
    }
#pragma unroll
    for (int j = 0; j < 4; ++j) {
      acc[0][j] = __builtin_amdgcn_wmma_f32_16x16x4_f32(false, a0, false, bb[j],
                                                        (short)0, acc[0][j], false, false);
      acc[1][j] = __builtin_amdgcn_wmma_f32_16x16x4_f32(false, a1, false, bb[j],
                                                        (short)0, acc[1][j], false, false);
    }
  }
#pragma unroll
  for (int i = 0; i < 2; ++i)
#pragma unroll
    for (int j = 0; j < 4; ++j)
      store_tile((bhalf*)nullptr, C, N, row0 + 16 * i, col0 + 16 * j, acc[i][j], scale, 0);
#else
  int lane = threadIdx.x & 31;
  int r = row0 + lane;
  for (int c = col0; c < col0 + 64; ++c) {
    float s = 0.f;
    for (int k = 0; k < K; ++k) s += A[(size_t)r * K + k] * B[(size_t)k * N + c];
    C[(size_t)r * N + c] = s * scale;
  }
#endif
}

// ---------------------------------------------------------------------------
// reductions (wave32)
// ---------------------------------------------------------------------------
__device__ __forceinline__ float block_reduce(float v, float* sh, int opmax) {
  int lane = threadIdx.x & 31, w = threadIdx.x >> 5, nw = blockDim.x >> 5;
#pragma unroll
  for (int o = 16; o > 0; o >>= 1) {
    float t = __shfl_xor(v, o, 32);
    v = opmax ? fmaxf(v, t) : v + t;
  }
  if (lane == 0) sh[w] = v;
  __syncthreads();
  v = (threadIdx.x < (unsigned)nw) ? sh[threadIdx.x] : (opmax ? -3.4e38f : 0.f);
  if (w == 0) {
#pragma unroll
    for (int o = 16; o > 0; o >>= 1) {
      float t = __shfl_xor(v, o, 32);
      v = opmax ? fmaxf(v, t) : v + t;
    }
    if (lane == 0) sh[0] = v;
  }
  __syncthreads();
  float r = sh[0];
  __syncthreads();
  return r;
}

__global__ void layernorm_kernel(const float* x, const float* w, const float* b, bhalf* xn) {
  __shared__ float sh[16];
  size_t row = blockIdx.x;
  const float* p = x + row * DMODEL;
  float s = 0.f, s2 = 0.f;
  for (int i = threadIdx.x; i < DMODEL; i += blockDim.x) { float v = p[i]; s += v; s2 += v * v; }
  s  = block_reduce(s,  sh, 0);
  s2 = block_reduce(s2, sh, 0);
  float mu = s * (1.f / DMODEL);
  float var = s2 * (1.f / DMODEL) - mu * mu;
  float rs = rsqrtf(var + 1e-5f);
  bhalf* o = xn + row * DMODEL;
  for (int i = threadIdx.x; i < DMODEL; i += blockDim.x)
    o[i] = (bhalf)((p[i] - mu) * rs * w[i] + b[i]);
}

template <typename T>
__global__ void softmax_rows(T* data, int C) {
  __shared__ float sh[16];
  size_t row = blockIdx.x;
  T* p = data + row * (size_t)C;
  float mx = -3.4e38f;
  for (int i = threadIdx.x; i < C; i += blockDim.x) mx = fmaxf(mx, (float)p[i]);
  mx = block_reduce(mx, sh, 1);
  float s = 0.f;
  for (int i = threadIdx.x; i < C; i += blockDim.x) s += __expf((float)p[i] - mx);
  s = block_reduce(s, sh, 0);
  float inv = 1.f / s;
  for (int i = threadIdx.x; i < C; i += blockDim.x)
    p[i] = (T)(__expf((float)p[i] - mx) * inv);
}

// ---------------------------------------------------------------------------
// misc elementwise kernels
// ---------------------------------------------------------------------------
__global__ void f32_to_bf16_kernel(const float* in, bhalf* out, int n) {
  int i = blockIdx.x * blockDim.x + threadIdx.x;
  if (i < n) out[i] = (bhalf)in[i];
}

__global__ void split_qkv_kernel(const bhalf* qkv, bhalf* q, bhalf* k, bhalf* v, bhalf* vT) {
  size_t idx = blockIdx.x * (size_t)blockDim.x + threadIdx.x;  // < NT*DMODEL
  int d = idx & 63; size_t t = idx >> 6;
  int hh = t & 7; t >>= 3;
  size_t nn = t & (SEQ - 1); size_t bb = t >> 12;
  size_t rowbase = (bb * SEQ + nn) * 1536;
  size_t e = (size_t)hh * DH + d;
  float qv = (float)qkv[rowbase + e];
  float kv = (float)qkv[rowbase + 512 + e];
  float vv = (float)qkv[rowbase + 1024 + e];
  size_t bh = bb * NH + hh;
  size_t o = (bh * SEQ + nn) * DH + d;
  q[o] = (bhalf)(qv * 0.125f);   // dh^-0.5
  k[o] = (bhalf)kv;
  v[o] = (bhalf)vv;
  vT[(bh * DH + d) * SEQ + nn] = (bhalf)vv;
}

__global__ void landmarks_kernel(const bhalf* q, const bhalf* k, bhalf* ql, bhalf* kl) {
  size_t idx = blockIdx.x * (size_t)blockDim.x + threadIdx.x;  // < 2*BH*NL*DH
  const bhalf* src; bhalf* dst;
  size_t j = idx;
  const size_t half = (size_t)BH * NL * DH;
  if (j >= half) { j -= half; src = k; dst = kl; } else { src = q; dst = ql; }
  int d = j & 63; size_t t = j >> 6;
  int mi = t & 255; size_t bh = t >> 8;
  const bhalf* p = src + (bh * SEQ + (size_t)mi * 16) * DH + d;
  float s = 0.f;
#pragma unroll
  for (int i = 0; i < 16; ++i) s += (float)p[i * DH];
  dst[(bh * NL + mi) * DH + d] = (bhalf)(s * (1.f / 16.f));
}

__global__ void init_scal_kernel(float* scal) {
  if (threadIdx.x == 0) { scal[0] = 0.f; scal[1] = 0.f; }
}

__global__ void rowabs_max_kernel(const float* a2, float* scal) {
  int row = blockIdx.x * (blockDim.x >> 5) + (threadIdx.x >> 5);  // < BH*NL
  int lane = threadIdx.x & 31;
  const float* p = a2 + (size_t)row * NL;
  float s = 0.f;
  for (int i = lane; i < NL; i += 32) s += fabsf(p[i]);
#pragma unroll
  for (int o = 16; o > 0; o >>= 1) s += __shfl_xor(s, o, 32);
  if (lane == 0) atomicMax((unsigned int*)scal, __float_as_uint(s));
}

__global__ void colabs_max_kernel(const float* a2, float* scal) {
  size_t idx = blockIdx.x * (size_t)blockDim.x + threadIdx.x;  // < BH*NL
  size_t bh = idx >> 8; int col = idx & 255;
  const float* p = a2 + bh * ((size_t)NL * NL) + col;
  float s = 0.f;
  for (int i = 0; i < NL; ++i) s += fabsf(p[(size_t)i * NL]);
  atomicMax((unsigned int*)(scal + 1), __float_as_uint(s));
}

__global__ void zinit_kernel(const float* a2, const float* scal, float* z) {
  size_t idx = blockIdx.x * (size_t)blockDim.x + threadIdx.x;  // < BH*NL*NL
  float inv = 1.f / (scal[0] * scal[1]);
  int j = idx & 255; size_t t = idx >> 8; int i = t & 255; size_t bh = t >> 8;
  z[(bh * NL + i) * NL + j] = a2[(bh * NL + j) * NL + i] * inv;  // a^T / scale
}

__global__ void axpbi_kernel(const float* in, float* out, float alpha, float beta) {
  size_t idx = blockIdx.x * (size_t)blockDim.x + threadIdx.x;  // < BH*NL*NL
  int j = idx & 255; int i = (idx >> 8) & 255;
  out[idx] = alpha * in[idx] + ((i == j) ? beta : 0.f);
}

__global__ void z_to_zT_kernel(const float* z, bhalf* zT) {
  size_t idx = blockIdx.x * (size_t)blockDim.x + threadIdx.x;  // < BH*NL*NL
  int j = idx & 255; int i = (idx >> 8) & 255; size_t bh = idx >> 16;
  zT[(bh << 16) + (size_t)j * NL + i] = (bhalf)z[idx];
}

__global__ void conv_merge_kernel(const float* outpre, const bhalf* v,
                                  const float* conv_w, bhalf* merged) {
  size_t idx = blockIdx.x * (size_t)blockDim.x + threadIdx.x;  // (bh, n, d) < BH*SEQ*DH
  int d = idx & 63; size_t t = idx >> 6;
  size_t nn = t & (SEQ - 1); size_t bh = t >> 12;
  int hh = (int)(bh & 7); size_t bb = bh >> 3;
  const float* w = conv_w + hh * KERN;
  float r = 0.f;
#pragma unroll
  for (int kk = 0; kk < KERN; ++kk) {
    long nk = (long)nn + kk - (KERN / 2);
    if (nk >= 0 && nk < SEQ)
      r += (float)v[(bh * SEQ + (size_t)nk) * DH + d] * w[kk];
  }
  float val = outpre[idx] + r;
  merged[(bb * SEQ + nn) * DMODEL + (size_t)hh * DH + d] = (bhalf)val;
}

__global__ void final_out_kernel(const float* x, const float* f, const float* out_b,
                                 const float* omega, float* out) {
  size_t idx = blockIdx.x * (size_t)blockDim.x + threadIdx.x;  // < NT*DMODEL
  int d = idx & (DMODEL - 1);
  out[idx] = x[idx] * omega[0] + f[idx] + out_b[d];
}

// ---------------------------------------------------------------------------
extern "C" void kernel_launch(void* const* d_in, const int* in_sizes, int n_in,
                              void* d_out, int out_size, void* d_ws, size_t ws_size,
                              hipStream_t stream) {
  (void)in_sizes; (void)n_in; (void)out_size; (void)ws_size;
  const float* x      = (const float*)d_in[0];
  const float* ln_w   = (const float*)d_in[1];
  const float* ln_b   = (const float*)d_in[2];
  const float* qkv_w  = (const float*)d_in[3];
  const float* out_w  = (const float*)d_in[4];
  const float* out_b  = (const float*)d_in[5];
  const float* conv_w = (const float*)d_in[6];
  const float* omega  = (const float*)d_in[7];
  float* out = (float*)d_out;

  size_t off = 0;
  auto alloc = [&](size_t bytes) -> void* {
    void* p = (char*)d_ws + off;
    off += (bytes + 255) & ~(size_t)255;
    return p;
  };
  bhalf* xn   = (bhalf*)alloc((size_t)NT * DMODEL * 2);
  bhalf* wqkv = (bhalf*)alloc((size_t)1536 * DMODEL * 2);
  bhalf* wout = (bhalf*)alloc((size_t)DMODEL * DMODEL * 2);
  bhalf* qkv  = (bhalf*)alloc((size_t)NT * 1536 * 2);       // freed after split -> reused for f
  bhalf* qb   = (bhalf*)alloc((size_t)BH * SEQ * DH * 2);
  bhalf* kb   = (bhalf*)alloc((size_t)BH * SEQ * DH * 2);
  bhalf* vb   = (bhalf*)alloc((size_t)BH * SEQ * DH * 2);
  bhalf* vT   = (bhalf*)alloc((size_t)BH * DH * SEQ * 2);
  bhalf* ql   = (bhalf*)alloc((size_t)BH * NL * DH * 2);
  bhalf* kl   = (bhalf*)alloc((size_t)BH * NL * DH * 2);
  bhalf* sim1 = (bhalf*)alloc((size_t)BH * SEQ * NL * 2);   // attn1; reused for merged
  bhalf* sim3 = (bhalf*)alloc((size_t)BH * NL * SEQ * 2);   // attn3; reused for outpre
  float* a2   = (float*)alloc((size_t)BH * NL * NL * 4);
  float* zA   = (float*)alloc((size_t)BH * NL * NL * 4);
  float* zB   = (float*)alloc((size_t)BH * NL * NL * 4);
  float* az   = (float*)alloc((size_t)BH * NL * NL * 4);
  float* tb   = (float*)alloc((size_t)BH * NL * NL * 4);
  float* ub   = (float*)alloc((size_t)BH * NL * NL * 4);
  bhalf* zT   = (bhalf*)alloc((size_t)BH * NL * NL * 2);
  bhalf* Pm   = (bhalf*)alloc((size_t)BH * SEQ * NL * 2);
  bhalf* QT   = (bhalf*)alloc((size_t)BH * DH * NL * 2);
  float* scal = (float*)alloc(8);
  // aliases into freed regions
  float* outpre = (float*)sim3;   // 67MB within 134MB (attn3 dead after QT gemm)
  bhalf* merged = sim1;           // 33.5MB within 134MB (attn1 dead after P gemm)
  float* fbuf   = (float*)qkv;    // 67MB within 100MB (qkv dead after split)

  auto gemm = [&](const bhalf* A, long long sA, const bhalf* Bt, long long sB,
                  bhalf* Cb, float* Cf, long long sC, int M, int N, int K,
                  int transC, float scale, int batch) {
    dim3 g(M / 256, N / 64, batch);
    gemm_bf16_abT<<<g, 128, 0, stream>>>(A, sA, Bt, sB, Cb, Cf, sC, M, N, K, transC, scale);
  };

  // 1) weights -> bf16
  f32_to_bf16_kernel<<<(1536 * 512 + 255) / 256, 256, 0, stream>>>(qkv_w, wqkv, 1536 * 512);
  f32_to_bf16_kernel<<<(512 * 512 + 255) / 256, 256, 0, stream>>>(out_w, wout, 512 * 512);

  // 2) layernorm -> xn (bf16)
  layernorm_kernel<<<NT, 128, 0, stream>>>(x, ln_w, ln_b, xn);

  // 3) qkv = xn @ Wqkv^T   [32768 x 1536]
  gemm(xn, 0, wqkv, 0, qkv, nullptr, 0, NT, 1536, DMODEL, 0, 1.f, 1);

  // 4) split into q (scaled), k, v, vT
  split_qkv_kernel<<<(NT * DMODEL) / 256, 256, 0, stream>>>(qkv, qb, kb, vb, vT);

  // 5) landmarks
  landmarks_kernel<<<(2 * BH * NL * DH) / 256, 256, 0, stream>>>(qb, kb, ql, kl);

  // 6) sim1 = q @ kl^T  [bh: 4096 x 256], K=64
  gemm(qb, (long long)SEQ * DH, kl, (long long)NL * DH, sim1, nullptr,
       (long long)SEQ * NL, SEQ, NL, DH, 0, 1.f, BH);
  // 7) sim2 = ql @ kl^T -> fp32 a2  [256 x 256]
  gemm(ql, (long long)NL * DH, kl, (long long)NL * DH, nullptr, a2,
       (long long)NL * NL, NL, NL, DH, 0, 1.f, BH);
  // 8) sim3 = ql @ k^T  [256 x 4096]
  gemm(ql, (long long)NL * DH, kb, (long long)SEQ * DH, sim3, nullptr,
       (long long)NL * SEQ, NL, SEQ, DH, 0, 1.f, BH);

  // 9-11) softmaxes
  softmax_rows<bhalf><<<BH * SEQ, 128, 0, stream>>>(sim1, NL);
  softmax_rows<float><<<BH * NL, 128, 0, stream>>>(a2, NL);
  softmax_rows<bhalf><<<BH * NL, 256, 0, stream>>>(sim3, SEQ);

  // 12-15) pinv init: scale = max(rowsum|a|) * max(colsum|a|); z = a^T/scale
  init_scal_kernel<<<1, 32, 0, stream>>>(scal);
  rowabs_max_kernel<<<(BH * NL) / 4, 128, 0, stream>>>(a2, scal);
  colabs_max_kernel<<<(BH * NL) / 256, 256, 0, stream>>>(a2, scal);
  zinit_kernel<<<(BH * NL * NL) / 256, 256, 0, stream>>>(a2, scal, zA);

  // 16) 6 Newton-Schulz iterations (fp32 WMMA)
  dim3 gp(NL / 128, NL / 64, BH);
  int ne = (BH * NL * NL) / 256;
  float* zcur = zA; float* zalt = zB;
  for (int it = 0; it < 6; ++it) {
    gemm_f32_ab<<<gp, 128, 0, stream>>>(a2, zcur, az, NL, NL, NL, 1.f);      // az = a@z
    axpbi_kernel<<<ne, 256, 0, stream>>>(az, tb, -1.f, 7.f);                 // t = 7I - az
    gemm_f32_ab<<<gp, 128, 0, stream>>>(az, tb, ub, NL, NL, NL, 1.f);        // u = az@t
    axpbi_kernel<<<ne, 256, 0, stream>>>(ub, tb, -1.f, 15.f);                // t = 15I - u
    gemm_f32_ab<<<gp, 128, 0, stream>>>(az, tb, ub, NL, NL, NL, 1.f);        // u = az@t
    axpbi_kernel<<<ne, 256, 0, stream>>>(ub, tb, -1.f, 13.f);                // t = 13I - u
    gemm_f32_ab<<<gp, 128, 0, stream>>>(zcur, tb, zalt, NL, NL, NL, 0.25f);  // z' = .25 z@t
    float* tmp = zcur; zcur = zalt; zalt = tmp;
  }

  // 17) zT (bf16)
  z_to_zT_kernel<<<ne, 256, 0, stream>>>(zcur, zT);

  // 18) P = attn1 @ z   [4096 x 256], K=256  (Bt = zT)
  gemm(sim1, (long long)SEQ * NL, zT, (long long)NL * NL, Pm, nullptr,
       (long long)SEQ * NL, SEQ, NL, NL, 0, 1.f, BH);
  // 19) QT = (attn3 @ v)^T  [64 x 256]  (Bt = vT, transC store)
  gemm(sim3, (long long)NL * SEQ, vT, (long long)DH * SEQ, QT, nullptr,
       (long long)DH * NL, NL, DH, SEQ, 1, 1.f, BH);
  // 20) outpre = P @ Q   [4096 x 64], K=256  (Bt = QT), fp32
  gemm(Pm, (long long)SEQ * NL, QT, (long long)DH * NL, nullptr, outpre,
       (long long)SEQ * DH, SEQ, DH, NL, 0, 1.f, BH);

  // 21) + depthwise conv residual, head-merge -> bf16 [b, n, 512]
  conv_merge_kernel<<<(BH * SEQ * DH) / 256, 256, 0, stream>>>(outpre, vb, conv_w, merged);

  // 22) f = merged @ Wout^T  [32768 x 512]
  gemm(merged, 0, wout, 0, nullptr, fbuf, 0, NT, DMODEL, DMODEL, 0, 1.f, 1);

  // 23) out = x*omega + f + out_b
  final_out_kernel<<<(NT * DMODEL) / 256, 256, 0, stream>>>(x, fbuf, out_b, omega, out);
}
